// AttentionModel_39848706573207
// MI455X (gfx1250) — compile-verified, matrix operand order checked by reading, not run on hardware
//
#include <hip/hip_runtime.h>
#include <math.h>

// Problem constants from the reference
#define B_DIM 4
#define S_DIM 4096
#define E_DIM 2048

// Tile sizes: block tile 128(M) x 64(N), K-stage 64 (2 WMMA k-steps), double-buffered
#define MT 128
#define NT 64
#define KT 64
#define PAD 8

typedef __attribute__((ext_vector_type(16))) __bf16 v16bf;
typedef __attribute__((ext_vector_type(8)))  float  v8f;

struct alignas(16) U4 { unsigned int x, y, z, w; };
struct alignas(8)  U2 { unsigned int x, y; };
struct alignas(16) F4 { float a, b, c, d; };

// ---- CDNA5 async global->LDS copy (ASYNCcnt-tracked), with fallbacks -------
#if defined(__has_builtin)
#if __has_builtin(__builtin_amdgcn_global_load_async_to_lds_b128)
#define HAVE_ASYNC_CP 1
#endif
#endif

#if defined(HAVE_ASYNC_CP)
// Builtin signature (from hipcc diagnostic): params are pointers to
// int __attribute__((vector_size(16))) in AS1 (src) and AS3 (dst), plus
// two integer immediates (offset, cpol).
typedef int v4i_g __attribute__((vector_size(16)));
typedef __attribute__((address_space(1))) v4i_g* as1_v4i;
typedef __attribute__((address_space(3))) v4i_g* as3_v4i;
// 16-byte async copy: global (AS1) -> LDS (AS3). AS3 offset = low 32 bits of
// the generic LDS address.
#define CP16(g, l)                                                    \
  __builtin_amdgcn_global_load_async_to_lds_b128(                     \
      (as1_v4i)(size_t)(g), (as3_v4i)(unsigned int)(size_t)(l), 0, 0)
#if __has_builtin(__builtin_amdgcn_s_wait_asynccnt)
#define WAIT_ASYNC(n) __builtin_amdgcn_s_wait_asynccnt(n)
#else
#define WAIT_ASYNC(n) asm volatile("s_wait_asynccnt %0" ::"i"(n) : "memory")
#endif
#else
// Synchronous fallback: b128 load + b128 LDS store
#define CP16(g, l) (*(U4*)(l) = *(const U4*)(g))
#define WAIT_ASYNC(n) ((void)0)
#endif

__device__ __forceinline__ unsigned short f2bf(float f) {
  union { float f; unsigned int u; } v; v.f = f;
  unsigned int u = v.u;
  u += 0x7fffu + ((u >> 16) & 1u);   // round-to-nearest-even
  return (unsigned short)(u >> 16);
}

// ---------------------------------------------------------------------------
// fp32 -> bf16 conversion (vectorized, grid-stride)
// ---------------------------------------------------------------------------
__global__ __launch_bounds__(256) void cvt_f32_bf16(const F4* __restrict__ src,
                                                    U2* __restrict__ dst,
                                                    long n4) {
  long i = (long)blockIdx.x * 256 + threadIdx.x;
  const long stride = (long)gridDim.x * 256;
  for (; i < n4; i += stride) {
    F4 f = src[i];
    U2 o;
    o.x = (unsigned)f2bf(f.a) | ((unsigned)f2bf(f.b) << 16);
    o.y = (unsigned)f2bf(f.c) | ((unsigned)f2bf(f.d) << 16);
    dst[i] = o;
  }
}

// ---------------------------------------------------------------------------
// bf16 WMMA GEMM, NT form:  C[m,n] = scale * sum_k A[m,k] * Bt[n,k]  (+bias[m])
//   A  : [M x K] row-major bf16
//   Bt : [N x K] row-major bf16
//   C  : bf16 or fp32; TRANS_C stores C[n,m] instead (used to produce V^T).
// Block: 256 threads = 8 waves; block tile 128x64; wave tile 32x32 (2x2 WMMA).
// Double-buffered LDS fed by async global->LDS b128 copies.
// ---------------------------------------------------------------------------
template<bool OUT_BF16, bool HAS_BIAS, bool TRANS_C>
__global__ __launch_bounds__(256) void wmma_gemm_nt(
    const unsigned short* __restrict__ A,  int lda, size_t strideA,
    const unsigned short* __restrict__ Bt, int ldb, size_t strideB,
    const float* __restrict__ bias,
    void* __restrict__ Cv, int ldc, size_t strideC,
    int K, float scale)
{
  __shared__ unsigned short As[2][MT][KT + PAD];  // 2 x 128 x 72 bf16 = 36 KB
  __shared__ unsigned short Bs[2][NT][KT + PAD];  // 2 x  64 x 72 bf16 = 18 KB

  const int tid  = threadIdx.x;
  const int lane = tid & 31;
  const int wave = tid >> 5;
  const int wm   = (wave & 3) * 32;   // 4 waves tile M (128)
  const int wn   = (wave >> 2) * 32;  // 2 waves tile N (64)
  const int hf   = lane >> 4;         // lane half (0/1)
  const int lrow = lane & 15;

  const int m0 = blockIdx.y * MT;
  const int n0 = blockIdx.x * NT;
  const int zb = blockIdx.z;

  const unsigned short* Ab = A  + (size_t)zb * strideA;
  const unsigned short* Bb = Bt + (size_t)zb * strideB;

  const v8f vzero = {0.f, 0.f, 0.f, 0.f, 0.f, 0.f, 0.f, 0.f};
  v8f acc[2][2];
  acc[0][0] = vzero; acc[0][1] = vzero; acc[1][0] = vzero; acc[1][1] = vzero;

  // Per-thread global->LDS fill coordinates (all aligned b128)
  const int arow = tid >> 1;          // 0..127, A: 2 threads/row, 4x16B each
  const int acol = (tid & 1) * 32;
  const int brow = tid >> 2;          // 0..63,  B: 4 threads/row, 2x16B each
  const int bcol = (tid & 3) * 16;

  auto fill = [&](int kk, int p) {
    const unsigned short* ga = Ab + (size_t)(m0 + arow) * lda + kk + acol;
    unsigned short* la = &As[p][arow][acol];
    CP16(ga,      la);
    CP16(ga + 8,  la + 8);
    CP16(ga + 16, la + 16);
    CP16(ga + 24, la + 24);
    const unsigned short* gb = Bb + (size_t)(n0 + brow) * ldb + kk + bcol;
    unsigned short* lb = &Bs[p][brow][bcol];
    CP16(gb,      lb);
    CP16(gb + 8,  lb + 8);
  };

  const int nk = K / KT;
  fill(0, 0);                         // prologue: stage tile 0

  union Frag { v16bf bf; U4 u[2]; };

  for (int i = 0; i < nk; ++i) {
    const int p = i & 1;
    if (i + 1 < nk) {
      fill((i + 1) * KT, 1 - p);      // overwrites buffer consumed at iter i-1
      WAIT_ASYNC(6);                  // in-order: tile i's 6 copies complete
    } else {
      WAIT_ASYNC(0);
    }
    __syncthreads();                  // tile i visible workgroup-wide

    #pragma unroll
    for (int s = 0; s < 2; ++s) {     // two 32-deep WMMA k-steps per stage
      Frag afrag[2], bfrag[2];
      #pragma unroll
      for (int ii = 0; ii < 2; ++ii) {
        // A 16x32 frag: lane m=lane&15; lanes 0-15 hold K 0-7,16-23;
        // lanes 16-31 hold K 8-15,24-31 (ISA 7.12.2)
        const unsigned short* pa = &As[p][wm + 16 * ii + lrow][32 * s];
        afrag[ii].u[0] = *(const U4*)(pa + 8 * hf);
        afrag[ii].u[1] = *(const U4*)(pa + 16 + 8 * hf);
      }
      #pragma unroll
      for (int jj = 0; jj < 2; ++jj) {
        // B 32x16 frag, mirrored layout from the [N][K] row-major tile
        const unsigned short* pb = &Bs[p][wn + 16 * jj + lrow][32 * s];
        bfrag[jj].u[0] = *(const U4*)(pb + 8 * hf);
        bfrag[jj].u[1] = *(const U4*)(pb + 16 + 8 * hf);
      }
      if (s == 1) __syncthreads();    // frags in regs; free buffer p for refill

      #pragma unroll
      for (int ii = 0; ii < 2; ++ii)
        #pragma unroll
        for (int jj = 0; jj < 2; ++jj)
          acc[ii][jj] = __builtin_amdgcn_wmma_f32_16x16x32_bf16(
              /*neg_a=*/false, afrag[ii].bf, /*neg_b=*/false, bfrag[jj].bf,
              /*c_mod=*/(short)0, acc[ii][jj],
              /*reuse_a=*/false, /*reuse_b=*/false);
    }
  }

  // Epilogue: C/D layout = VGPR r, lanes 0-15: M=r,N=lane; lanes 16-31: M=8+r
  #pragma unroll
  for (int ii = 0; ii < 2; ++ii) {
    #pragma unroll
    for (int jj = 0; jj < 2; ++jj) {
      #pragma unroll
      for (int r = 0; r < 8; ++r) {
        const int mg = m0 + wm + 16 * ii + r + 8 * hf;
        const int ng = n0 + wn + 16 * jj + lrow;
        float val = acc[ii][jj][r] * scale;
        if (HAS_BIAS) val += bias[mg];
        const size_t idx = TRANS_C
            ? (size_t)zb * strideC + (size_t)ng * ldc + mg
            : (size_t)zb * strideC + (size_t)mg * ldc + ng;
        if (OUT_BF16) ((unsigned short*)Cv)[idx] = f2bf(val);
        else          ((float*)Cv)[idx] = val;
      }
    }
  }
}

// ---------------------------------------------------------------------------
// Row softmax: one block per row of length E_DIM (fp32 in, bf16 out)
// ---------------------------------------------------------------------------
__global__ __launch_bounds__(256) void softmax_rows(const float* __restrict__ scores,
                                                    unsigned short* __restrict__ attn) {
  __shared__ float red[256];
  const int row = blockIdx.x;                       // b*E + e
  const float* src = scores + (size_t)row * E_DIM;
  unsigned short* dst = attn + (size_t)row * E_DIM;
  const int tid = threadIdx.x;

  float vals[8];
  float m = -3.402823466e38f;
  #pragma unroll
  for (int i = 0; i < 8; ++i) {
    vals[i] = src[tid + i * 256];
    m = fmaxf(m, vals[i]);
  }
  red[tid] = m; __syncthreads();
  for (int s = 128; s > 0; s >>= 1) {
    if (tid < s) red[tid] = fmaxf(red[tid], red[tid + s]);
    __syncthreads();
  }
  m = red[0]; __syncthreads();

  float sum = 0.f;
  #pragma unroll
  for (int i = 0; i < 8; ++i) { vals[i] = __expf(vals[i] - m); sum += vals[i]; }
  red[tid] = sum; __syncthreads();
  for (int s = 128; s > 0; s >>= 1) {
    if (tid < s) red[tid] += red[tid + s];
    __syncthreads();
  }
  const float inv = 1.0f / red[0];

  #pragma unroll
  for (int i = 0; i < 8; ++i) dst[tid + i * 256] = f2bf(vals[i] * inv);
}

// ---------------------------------------------------------------------------
// Launch
// ---------------------------------------------------------------------------
extern "C" void kernel_launch(void* const* d_in, const int* in_sizes, int n_in,
                              void* d_out, int out_size, void* d_ws, size_t ws_size,
                              hipStream_t stream) {
  (void)in_sizes; (void)n_in; (void)out_size; (void)ws_size;

  const float* x  = (const float*)d_in[0];
  const float* Wq = (const float*)d_in[1];
  const float* bq = (const float*)d_in[2];
  const float* Wk = (const float*)d_in[3];
  const float* bk = (const float*)d_in[4];
  const float* Wv = (const float*)d_in[5];
  const float* bv = (const float*)d_in[6];

  const size_t BSE = (size_t)B_DIM * S_DIM * E_DIM;
  const size_t EE  = (size_t)E_DIM * E_DIM;
  const size_t BEE = (size_t)B_DIM * EE;

  // Workspace layout (bf16 buffers stored as raw ushort)
  char* p = (char*)d_ws;
  unsigned short* xb     = (unsigned short*)p; p += BSE * 2;
  unsigned short* Wqb    = (unsigned short*)p; p += EE * 2;
  unsigned short* Wkb    = (unsigned short*)p; p += EE * 2;
  unsigned short* Wvb    = (unsigned short*)p; p += EE * 2;
  unsigned short* qb     = (unsigned short*)p; p += BSE * 2;  // [B][E][S]
  unsigned short* kb     = (unsigned short*)p; p += BSE * 2;  // [B][E][S]
  unsigned short* vtb    = (unsigned short*)p; p += BSE * 2;  // [B][S][E]  (V^T)
  float*          scores = (float*)p;          p += BEE * 4;  // [B][E][E]
  unsigned short* attn   = (unsigned short*)p;                // [B][E][E]

  // 1) fp32 -> bf16 conversions
  cvt_f32_bf16<<<4096, 256, 0, stream>>>((const F4*)x,  (U2*)xb,  (long)(BSE / 4));
  cvt_f32_bf16<<<512,  256, 0, stream>>>((const F4*)Wq, (U2*)Wqb, (long)(EE / 4));
  cvt_f32_bf16<<<512,  256, 0, stream>>>((const F4*)Wk, (U2*)Wkb, (long)(EE / 4));
  cvt_f32_bf16<<<512,  256, 0, stream>>>((const F4*)Wv, (U2*)Wvb, (long)(EE / 4));

  // 2) Projections: P[f,s] = sum_e W[f,e] * x[b,s,e] + b[f]
  //    A = W [E x E], Bt = x[b] [S x E], K = E. q,k stored [E][S]; v stored ^T [S][E].
  dim3 gq(S_DIM / NT, E_DIM / MT, B_DIM);
  wmma_gemm_nt<true, true, false><<<gq, 256, 0, stream>>>(
      Wqb, E_DIM, 0, xb, E_DIM, (size_t)S_DIM * E_DIM, bq,
      qb, S_DIM, (size_t)E_DIM * S_DIM, E_DIM, 1.0f);
  wmma_gemm_nt<true, true, false><<<gq, 256, 0, stream>>>(
      Wkb, E_DIM, 0, xb, E_DIM, (size_t)S_DIM * E_DIM, bk,
      kb, S_DIM, (size_t)E_DIM * S_DIM, E_DIM, 1.0f);
  wmma_gemm_nt<true, true, true><<<gq, 256, 0, stream>>>(
      Wvb, E_DIM, 0, xb, E_DIM, (size_t)S_DIM * E_DIM, bv,
      vtb, E_DIM, (size_t)E_DIM * S_DIM, E_DIM, 1.0f);

  // 3) scores[b,e,f] = (1/sqrt(E)) * sum_s q[b,e,s] * k[b,f,s]
  //    A = q [E x S], Bt = k [E x S], K = S; fp32 out
  const float inv_scale = 0.022097086912079608f;  // 1/sqrt(2048)
  dim3 gs(E_DIM / NT, E_DIM / MT, B_DIM);
  wmma_gemm_nt<false, false, false><<<gs, 256, 0, stream>>>(
      qb, S_DIM, (size_t)E_DIM * S_DIM, kb, S_DIM, (size_t)E_DIM * S_DIM, nullptr,
      scores, E_DIM, EE, S_DIM, inv_scale);

  // 4) softmax over f -> bf16 attn
  softmax_rows<<<B_DIM * E_DIM, 256, 0, stream>>>(scores, attn);

  // 5) out[b,e,s] = sum_f attn[b,e,f] * v[b,f,s] = sum_f attn[e,f] * vt[s,f]
  //    A = attn [E x E], Bt = vt [S x E], K = E; fp32 out to d_out
  dim3 go(S_DIM / NT, E_DIM / MT, B_DIM);
  wmma_gemm_nt<false, false, false><<<go, 256, 0, stream>>>(
      attn, E_DIM, EE, vtb, E_DIM, (size_t)S_DIM * E_DIM, nullptr,
      d_out, S_DIM, (size_t)E_DIM * S_DIM, E_DIM, 1.0f);
}
